// MultiSimilarityLoss_17729624998286
// MI455X (gfx1250) — matrix operand order repaired and checked by reading, not compiled, not run on hardware
//
#include <hip/hip_runtime.h>
#include <hip/hip_bf16.h>
#include <math.h>

typedef __attribute__((ext_vector_type(16))) _Float16 v16h;
typedef __attribute__((ext_vector_type(8)))  _Float16 v8h;
typedef __attribute__((ext_vector_type(4)))  _Float16 v4h;
typedef __attribute__((ext_vector_type(8)))  float    v8f;

#define BN 8192
#define DN 128
#define ONE_EPS 0.99999f   // 1 - 1e-5
#define MARGIN  0.3f
#define THRESH  0.5f

// ---------------------------------------------------------------------------
// Kernel 1: L2-normalize rows, emit f16.  One wave32 per row (4 floats/lane).
// ---------------------------------------------------------------------------
__global__ __launch_bounds__(256) void l2norm_f16_kernel(
    const float* __restrict__ feats, _Float16* __restrict__ fh) {
  const int row  = (blockIdx.x * 256 + threadIdx.x) >> 5;   // global wave = row
  const int lane = threadIdx.x & 31;
  const float4 v = ((const float4*)(feats + (size_t)row * DN))[lane];
  float ss = v.x * v.x + v.y * v.y + v.z * v.z + v.w * v.w;
#pragma unroll
  for (int m = 16; m >= 1; m >>= 1) ss += __shfl_xor(ss, m, 32);
  const float inv = rsqrtf(ss);
  v4h o;
  o[0] = (_Float16)(v.x * inv);
  o[1] = (_Float16)(v.y * inv);
  o[2] = (_Float16)(v.z * inv);
  o[3] = (_Float16)(v.w * inv);
  *(v4h*)(fh + (size_t)row * DN + lane * 4) = o;
}

// ---------------------------------------------------------------------------
// Kernel 2: two-pass WMMA sweep.  Block = 4 waves, owns 32 rows.
// Each wave holds TWO 16-row A stripes in VGPRs (64 regs) and produces two
// 16x16 output tiles per B fragment: 8 independent-chain WMMAs per 8 loads.
//   Pass 1: min over positives / max over negatives per row.
//   Pass 2: masked exp-sums, per-row loss + invalid flag.
// ---------------------------------------------------------------------------
__global__ __launch_bounds__(128) void msloss_kernel(
    const _Float16* __restrict__ fh, const int* __restrict__ lab,
    float* __restrict__ rowloss, float* __restrict__ rowinv) {
  __shared__ float sA[4][32], sB[4][32], sMin[32], sMax[32];

  const int tid  = threadIdx.x;
  const int wv   = tid >> 5;
  const int lane = tid & 31;
  const int hi   = lane >> 4;   // half-wave select
  const int lo   = lane & 15;
  const int r0   = blockIdx.x << 5;          // 32 rows per block

  // --- A fragments: two row stripes, all K=128, resident in VGPRs.
  // Layout (ISA 7.12.2, 16-bit A 16x32): lanes 0-15 row=lo K 0-7/16-23,
  // lanes 16-31 same rows K 8-15/24-31.
  v16h afrag[2][4];
#pragma unroll
  for (int t = 0; t < 2; ++t) {
    const _Float16* arow = fh + (size_t)(r0 + 16 * t + lo) * DN + 8 * hi;
#pragma unroll
    for (int k = 0; k < 4; ++k) {
      v8h* ap = reinterpret_cast<v8h*>(&afrag[t][k]);
      ap[0] = *(const v8h*)(arow + 32 * k);
      ap[1] = *(const v8h*)(arow + 32 * k + 16);
    }
  }
  // Row labels: C/D layout puts row m = j + 8*hi in VGPR j.
  int rl[2][8];
#pragma unroll
  for (int t = 0; t < 2; ++t)
#pragma unroll
    for (int j = 0; j < 8; ++j) rl[t][j] = lab[r0 + 16 * t + 8 * hi + j];

  // ------------------------------- pass 1 ---------------------------------
  float minp[2][8], maxn[2][8];
#pragma unroll
  for (int t = 0; t < 2; ++t)
#pragma unroll
    for (int j = 0; j < 8; ++j) { minp[t][j] = INFINITY; maxn[t][j] = -INFINITY; }

  for (int ct = wv; ct < BN / 16; ct += 4) {       // 128 uniform iters/wave
    const int c0i = ct << 4;
    // B layout (32x16 f16): lane = column, lanes 0-15 K 0-7/16-23,
    // lanes 16-31 K 8-15/24-31.
    const _Float16* brow = fh + (size_t)(c0i + lo) * DN + 8 * hi;
    v8f c0 = {}, c1 = {};
#pragma unroll
    for (int k = 0; k < 4; ++k) {
      v16h bfrag;
      v8h* bp = reinterpret_cast<v8h*>(&bfrag);
      bp[0] = *(const v8h*)(brow + 32 * k);
      bp[1] = *(const v8h*)(brow + 32 * k + 16);
      c0 = __builtin_amdgcn_wmma_f32_16x16x32_f16(
          false, afrag[0][k], false, bfrag, (short)0, c0, false, false);
      c1 = __builtin_amdgcn_wmma_f32_16x16x32_f16(
          false, afrag[1][k], false, bfrag, (short)0, c1, false, false);
    }
    const int cl = lab[c0i + lo];
#pragma unroll
    for (int j = 0; j < 8; ++j) {
      {
        const float s    = c0[j];
        const bool  same = (rl[0][j] == cl);
        minp[0][j] = (same && s < ONE_EPS) ? fminf(minp[0][j], s) : minp[0][j];
        maxn[0][j] = (!same)               ? fmaxf(maxn[0][j], s) : maxn[0][j];
      }
      {
        const float s    = c1[j];
        const bool  same = (rl[1][j] == cl);
        minp[1][j] = (same && s < ONE_EPS) ? fminf(minp[1][j], s) : minp[1][j];
        maxn[1][j] = (!same)               ? fmaxf(maxn[1][j], s) : maxn[1][j];
      }
    }
  }
  // reduce across the 16 lanes that share each row
#pragma unroll
  for (int m = 1; m < 16; m <<= 1) {
#pragma unroll
    for (int t = 0; t < 2; ++t)
#pragma unroll
      for (int j = 0; j < 8; ++j) {
        minp[t][j] = fminf(minp[t][j], __shfl_xor(minp[t][j], m, 32));
        maxn[t][j] = fmaxf(maxn[t][j], __shfl_xor(maxn[t][j], m, 32));
      }
  }
  if (lo == 0) {
#pragma unroll
    for (int t = 0; t < 2; ++t)
#pragma unroll
      for (int j = 0; j < 8; ++j) {
        sA[wv][16 * t + 8 * hi + j] = minp[t][j];
        sB[wv][16 * t + 8 * hi + j] = maxn[t][j];
      }
  }
  __syncthreads();
  if (tid < 32) {
    float mn = INFINITY, mx = -INFINITY;
#pragma unroll
    for (int w = 0; w < 4; ++w) {
      mn = fminf(mn, sA[w][tid]);
      mx = fmaxf(mx, sB[w][tid]);
    }
    sMin[tid] = mn;
    sMax[tid] = mx;
  }
  __syncthreads();
  float mp[2][8], mxn[2][8];
#pragma unroll
  for (int t = 0; t < 2; ++t)
#pragma unroll
    for (int j = 0; j < 8; ++j) {
      mp[t][j]  = sMin[16 * t + 8 * hi + j];
      mxn[t][j] = sMax[16 * t + 8 * hi + j];
    }
  __syncthreads();   // sA/sB reused below

  // ------------------------------- pass 2 ---------------------------------
  float ps[2][8], ns[2][8];
#pragma unroll
  for (int t = 0; t < 2; ++t)
#pragma unroll
    for (int j = 0; j < 8; ++j) { ps[t][j] = 0.0f; ns[t][j] = 0.0f; }

  for (int ct = wv; ct < BN / 16; ct += 4) {
    const int c0i = ct << 4;
    const _Float16* brow = fh + (size_t)(c0i + lo) * DN + 8 * hi;
    v8f c0 = {}, c1 = {};
#pragma unroll
    for (int k = 0; k < 4; ++k) {
      v16h bfrag;
      v8h* bp = reinterpret_cast<v8h*>(&bfrag);
      bp[0] = *(const v8h*)(brow + 32 * k);
      bp[1] = *(const v8h*)(brow + 32 * k + 16);
      c0 = __builtin_amdgcn_wmma_f32_16x16x32_f16(
          false, afrag[0][k], false, bfrag, (short)0, c0, false, false);
      c1 = __builtin_amdgcn_wmma_f32_16x16x32_f16(
          false, afrag[1][k], false, bfrag, (short)0, c1, false, false);
    }
    const int cl = lab[c0i + lo];
#pragma unroll
    for (int j = 0; j < 8; ++j) {
#pragma unroll
      for (int t = 0; t < 2; ++t) {
        const float s    = (t == 0) ? c0[j] : c1[j];
        const bool  same = (rl[t][j] == cl);
        const bool  posk = same && (s < ONE_EPS) && (s < mxn[t][j]);
        const bool  negk = (!same) && (s + MARGIN > mp[t][j]);
        ps[t][j] += posk ? __expf(-0.2f * (s - THRESH)) : 0.0f;
        ns[t][j] += negk ? __expf(40.0f * (s - THRESH)) : 0.0f;
      }
    }
  }
#pragma unroll
  for (int m = 1; m < 16; m <<= 1) {
#pragma unroll
    for (int t = 0; t < 2; ++t)
#pragma unroll
      for (int j = 0; j < 8; ++j) {
        ps[t][j] += __shfl_xor(ps[t][j], m, 32);
        ns[t][j] += __shfl_xor(ns[t][j], m, 32);
      }
  }
  if (lo == 0) {
#pragma unroll
    for (int t = 0; t < 2; ++t)
#pragma unroll
      for (int j = 0; j < 8; ++j) {
        sA[wv][16 * t + 8 * hi + j] = ps[t][j];
        sB[wv][16 * t + 8 * hi + j] = ns[t][j];
      }
  }
  __syncthreads();
  if (tid < 32) {
    float p = 0.0f, n = 0.0f;
#pragma unroll
    for (int w = 0; w < 4; ++w) { p += sA[w][tid]; n += sB[w][tid]; }
    const bool valid = (p > 0.0f) && (n > 0.0f);
    // 1/SCALE_POS = 5, 1/SCALE_NEG = 0.025
    rowloss[r0 + tid] = valid ? (log1pf(p) * 5.0f + log1pf(n) * 0.025f) : 0.0f;
    rowinv[r0 + tid]  = valid ? 0.0f : 1.0f;
  }
}

// ---------------------------------------------------------------------------
// Kernel 3: deterministic fixed-order final reduction.
// ---------------------------------------------------------------------------
__global__ __launch_bounds__(256) void finalize_kernel(
    const float* __restrict__ rowloss, const float* __restrict__ rowinv,
    float* __restrict__ out) {
  __shared__ float sl[256], si[256];
  const int t = threadIdx.x;
  float a = 0.0f, b = 0.0f;
  for (int i = t; i < BN; i += 256) { a += rowloss[i]; b += rowinv[i]; }
  sl[t] = a;
  si[t] = b;
  __syncthreads();
  for (int s = 128; s > 0; s >>= 1) {
    if (t < s) { sl[t] += sl[t + s]; si[t] += si[t + s]; }
    __syncthreads();
  }
  if (t == 0) {
    out[0] = sl[0] / (float)BN;   // loss
    out[1] = si[0];               // number of skipped samples
  }
}

// ---------------------------------------------------------------------------
extern "C" void kernel_launch(void* const* d_in, const int* in_sizes, int n_in,
                              void* d_out, int out_size, void* d_ws, size_t ws_size,
                              hipStream_t stream) {
  (void)in_sizes; (void)n_in; (void)out_size; (void)ws_size;
  const float* feats = (const float*)d_in[0];
  const int*   lab   = (const int*)d_in[1];

  _Float16* fh      = (_Float16*)d_ws;                       // 8192*128*2 = 2 MB
  float*    rowloss = (float*)((char*)d_ws + (size_t)BN * DN * sizeof(_Float16));
  float*    rowinv  = rowloss + BN;

  l2norm_f16_kernel<<<BN / 8, 256, 0, stream>>>(feats, fh);
  msloss_kernel<<<BN / 32, 128, 0, stream>>>(fh, lab, rowloss, rowinv);
  finalize_kernel<<<1, 256, 0, stream>>>(rowloss, rowinv, (float*)d_out);
}
